// Attention_73323681677530
// MI455X (gfx1250) — matrix operand order repaired and checked by reading, not compiled, not run on hardware
//
#include <hip/hip_runtime.h>
#include <hip/hip_bf16.h>

// ---------------------------------------------------------------------------
// CDNA5 WMMA:  v_wmma_f32_16x16x32_bf16
//   D(16x16 f32, 8 VGPR) = A(16x32 bf16, 8 VGPR) * B(32x16 bf16, 8 VGPR) + C
// ---------------------------------------------------------------------------
typedef __attribute__((ext_vector_type(16))) __bf16 bf16x16;
typedef __attribute__((ext_vector_type(8)))  __bf16 bf16x8;
typedef __attribute__((ext_vector_type(4)))  __bf16 bf16x4;
typedef __attribute__((ext_vector_type(4)))  float  fvec4;
typedef __attribute__((ext_vector_type(8)))  float  f32x8;

#define BM 128
#define BN 128
#define BK 32
#define LDSROW (BK + 8)   // 40 bf16 = 80B rows -> 16B k-chunks stay aligned

// Convert float4 -> bf16x4 (2x v_cvt_pk_bf16_f32) and store as one ds_store_b64.
__device__ __forceinline__
void cvt_store4(__bf16* dst, const float4& v)
{
    union { bf16x4 h; uint2 u; } c;
    c.h = __builtin_convertvector(*(const fvec4*)&v, bf16x4);
    *(uint2*)dst = c.u;
}

// ---- staging helpers (layout resolved at compile time) ---------------------
template<bool AKC>  // AKC: A k-contiguous (as_col==1)
__device__ __forceinline__
void load_a(const float* __restrict__ A, int as_row, int as_col, int bm, int tid,
            int kbase, float4 (&ra)[4], float (&ras)[16])
{
    if (AKC) {
        #pragma unroll
        for (int i = 0; i < 4; ++i) {
            const int f = tid + (i << 8);                       // 1024 float4s
            ra[i] = *(const float4*)(A + (bm + (f >> 3)) * as_row + kbase + ((f & 7) << 2));
        }
    } else {                                                    // m-fastest scalar (as_row==1)
        #pragma unroll
        for (int i = 0; i < 16; ++i) {
            const int e = tid + (i << 8);
            ras[i] = A[(bm + (e & 127)) * as_row + (kbase + (e >> 7)) * as_col];
        }
    }
}

template<bool AKC>
__device__ __forceinline__
void store_a(__bf16 (*As)[LDSROW], int tid, const float4 (&ra)[4], const float (&ras)[16])
{
    if (AKC) {
        #pragma unroll
        for (int i = 0; i < 4; ++i) {
            const int f = tid + (i << 8);
            cvt_store4(&As[f >> 3][(f & 7) << 2], ra[i]);
        }
    } else {
        #pragma unroll
        for (int i = 0; i < 16; ++i) {
            const int e = tid + (i << 8);
            As[e & 127][e >> 7] = (__bf16)ras[i];
        }
    }
}

template<bool BKC>  // BKC: B k-contiguous (bs_k==1, "NT"); else NN (bs_n==1)
__device__ __forceinline__
void load_b(const float* __restrict__ B, int bs_n, int bs_k, int bn, int tid,
            int kbase, float4 (&rb)[4])
{
    if (BKC) {
        #pragma unroll
        for (int i = 0; i < 4; ++i) {
            const int f = tid + (i << 8);                       // [n][k] float4 along k
            rb[i] = *(const float4*)(B + (bn + (f >> 3)) * bs_n + kbase + ((f & 7) << 2));
        }
    } else {
        #pragma unroll
        for (int i = 0; i < 4; ++i) {
            const int f = tid + (i << 8);                       // [k][n] float4 along n
            rb[i] = *(const float4*)(B + (kbase + (f >> 5)) * bs_k + bn + ((f & 31) << 2));
        }
    }
}

template<bool BKC>
__device__ __forceinline__
void store_b(__bf16 (*Bs)[LDSROW], int tid, const float4 (&rb)[4])
{
    if (BKC) {
        #pragma unroll
        for (int i = 0; i < 4; ++i) {
            const int f = tid + (i << 8);
            cvt_store4(&Bs[f >> 3][(f & 7) << 2], rb[i]);
        }
    } else {                                                    // transpose into Bs[n][k]
        #pragma unroll
        for (int i = 0; i < 4; ++i) {
            const int f = tid + (i << 8);
            const int kk = f >> 5, n4 = (f & 31) << 2;
            const bf16x4 h = __builtin_convertvector(*(const fvec4*)&rb[i], bf16x4);
            Bs[n4 + 0][kk] = h[0];
            Bs[n4 + 1][kk] = h[1];
            Bs[n4 + 2][kk] = h[2];
            Bs[n4 + 3][kk] = h[3];
        }
    }
}

// ---------------------------------------------------------------------------
// GEMM:  C[z][m][n] = sum_k A[z][m][k]*B[z][k][n] (+bias[n]),  f32 in/out,
// bf16 WMMA / f32 accumulate, LDS ping-pong double buffer, 1 barrier per tile.
// Requires M%128==0, N%128==0, K%32==0.
// ---------------------------------------------------------------------------
template<bool AKC, bool BKC>
__global__ __launch_bounds__(256)
void gemm_bf16wmma(const float* __restrict__ A, int as_row, int as_col, long long a_batch,
                   const float* __restrict__ B, int bs_n, int bs_k, long long b_batch,
                   float* __restrict__ C, long long c_batch,
                   const float* __restrict__ bias,
                   int N, int K)
{
    __shared__ alignas(16) __bf16 As[2][BM][LDSROW];   // [buf][m][k]
    __shared__ alignas(16) __bf16 Bs[2][BN][LDSROW];   // [buf][n][k]

    const int tid  = threadIdx.x;
    const int lane = tid & 31;
    const int wid  = tid >> 5;
    const int wm   = wid & 3;        // 4 waves x 32 rows
    const int wn   = wid >> 2;       // 2 waves x 64 cols
    const int bm   = blockIdx.y * BM;
    const int bn   = blockIdx.x * BN;

    A += (long long)blockIdx.z * a_batch;
    B += (long long)blockIdx.z * b_batch;
    C += (long long)blockIdx.z * c_batch;

    f32x8 acc[2][4];
    #pragma unroll
    for (int i = 0; i < 2; ++i)
        #pragma unroll
        for (int j = 0; j < 4; ++j)
            #pragma unroll
            for (int e = 0; e < 8; ++e) acc[i][j][e] = 0.0f;

    const int khalf = (lane >> 4) << 3;   // A frag: lanes 0-15 K 0..7/16..23, lanes 16-31 K 8..15/24..31
    const int kb16  = (lane >> 4) << 4;   // unused; B frag reads khalf, same k order as A
    const int rl    = lane & 15;

    float4 ra[4]; float ras[16]; float4 rb[4];

    // prologue: tile 0 -> regs -> LDS buf 0
    load_a<AKC>(A, as_row, as_col, bm, tid, 0, ra, ras);
    load_b<BKC>(B, bs_n, bs_k, bn, tid, 0, rb);
    store_a<AKC>(As[0], tid, ra, ras);
    store_b<BKC>(Bs[0], tid, rb);
    __syncthreads();

    int cur = 0;
    for (int k0 = 0; k0 < K; k0 += BK) {
        const bool more = (k0 + BK) < K;
        if (more) {                                      // issue next tile's global loads
            load_a<AKC>(A, as_row, as_col, bm, tid, k0 + BK, ra, ras);
            load_b<BKC>(B, bs_n, bs_k, bn, tid, k0 + BK, rb);
            if (AKC && (k0 + 2 * BK < K))                // L2 prefetch of tile k+2
                __builtin_prefetch(A + (bm + (tid >> 3)) * as_row + k0 + 2 * BK, 0, 1);
        }

        // compute tile k0 from buf[cur] (overlaps the in-flight global loads)
        bf16x16 afrag[2];
        #pragma unroll
        for (int mt = 0; mt < 2; ++mt) {
            const int r = wm * 32 + mt * 16 + rl;
            bf16x8 lo = *(const bf16x8*)&As[cur][r][khalf];
            bf16x8 hi = *(const bf16x8*)&As[cur][r][16 + khalf];
            afrag[mt] = __builtin_shufflevector(lo, hi, 0,1,2,3,4,5,6,7,8,9,10,11,12,13,14,15);
        }
        bf16x16 bfrag[4];
        #pragma unroll
        for (int nt = 0; nt < 4; ++nt) {
            const int n = wn * 64 + nt * 16 + rl;
            bf16x8 lo = *(const bf16x8*)&Bs[cur][n][khalf];
            bf16x8 hi = *(const bf16x8*)&Bs[cur][n][16 + khalf];
            bfrag[nt] = __builtin_shufflevector(lo, hi, 0,1,2,3,4,5,6,7,8,9,10,11,12,13,14,15);
        }
        #pragma unroll
        for (int mt = 0; mt < 2; ++mt)
            #pragma unroll
            for (int nt = 0; nt < 4; ++nt)
                acc[mt][nt] = __builtin_amdgcn_wmma_f32_16x16x32_bf16(
                    false, afrag[mt], false, bfrag[nt],
                    (short)0, acc[mt][nt], false, false);

        if (more) {                                      // stage next tile -> other buffer
            store_a<AKC>(As[cur ^ 1], tid, ra, ras);
            store_b<BKC>(Bs[cur ^ 1], tid, rb);
            __syncthreads();                             // single barrier per iteration
            cur ^= 1;
        }
    }

    // C/D layout: VGPR v -> row (lane>>4)*8 + v, col = lane&15
    const int r0 = (lane >> 4) * 8;
    #pragma unroll
    for (int mt = 0; mt < 2; ++mt) {
        #pragma unroll
        for (int nt = 0; nt < 4; ++nt) {
            const int col = bn + wn * 64 + nt * 16 + rl;
            const float bv = bias ? bias[col] : 0.0f;
            #pragma unroll
            for (int v = 0; v < 8; ++v) {
                const int row = bm + wm * 32 + mt * 16 + r0 + v;
                C[(long long)row * N + col] = acc[mt][nt][v] + bv;
            }
        }
    }
}

// ---------------------------------------------------------------------------
// In-place row softmax, ncols must be 2048 (row lives in registers: 2x float4).
// ---------------------------------------------------------------------------
__global__ __launch_bounds__(256)
void softmax_rows(float* __restrict__ p, int ncols)
{
    float4* row = (float4*)(p + (long long)blockIdx.x * ncols);
    __shared__ float red[256];
    const int t = threadIdx.x;

    float4 v[2];
    float m = -3.402823466e+38f;
    #pragma unroll
    for (int i = 0; i < 2; ++i) {
        v[i] = row[t + (i << 8)];
        m = fmaxf(m, fmaxf(fmaxf(v[i].x, v[i].y), fmaxf(v[i].z, v[i].w)));
    }
    red[t] = m; __syncthreads();
    for (int s = 128; s > 0; s >>= 1) {
        if (t < s) red[t] = fmaxf(red[t], red[t + s]);
        __syncthreads();
    }
    m = red[0]; __syncthreads();

    float sum = 0.0f;
    #pragma unroll
    for (int i = 0; i < 2; ++i) {
        v[i].x = __expf(v[i].x - m); v[i].y = __expf(v[i].y - m);
        v[i].z = __expf(v[i].z - m); v[i].w = __expf(v[i].w - m);
        sum += (v[i].x + v[i].y) + (v[i].z + v[i].w);
    }
    red[t] = sum; __syncthreads();
    for (int s = 128; s > 0; s >>= 1) {
        if (t < s) red[t] += red[t + s];
        __syncthreads();
    }
    const float inv = 1.0f / red[0];
    #pragma unroll
    for (int i = 0; i < 2; ++i) {
        v[i].x *= inv; v[i].y *= inv; v[i].z *= inv; v[i].w *= inv;
        row[t + (i << 8)] = v[i];
    }
}

// bc[e] = sum_t bq[t] * W_bil[t][e]   (folded bilinear bias)
__global__ __launch_bounds__(256)
void bil_bias(const float* __restrict__ bq, const float* __restrict__ Wb,
              float* __restrict__ bc, int D)
{
    const int e = blockIdx.x * 256 + threadIdx.x;
    float s = 0.0f;
    for (int tt = 0; tt < D; ++tt) s += bq[tt] * Wb[(long long)tt * D + e];
    bc[e] = s;
}

// ---------------------------------------------------------------------------
template<bool AKC, bool BKC>
static void launch_gemm(const float* A, int asr, int asc, long long ab,
                        const float* B, int bsn, int bsk, long long bb,
                        float* C, long long cb, const float* bias,
                        int M, int N, int K, int batch, hipStream_t s)
{
    dim3 grid((unsigned)(N / BN), (unsigned)(M / BM), (unsigned)batch);
    gemm_bf16wmma<AKC, BKC><<<grid, 256, 0, s>>>(A, asr, asc, ab, B, bsn, bsk, bb,
                                                 C, cb, bias, N, K);
}

extern "C" void kernel_launch(void* const* d_in, const int* in_sizes, int n_in,
                              void* d_out, int out_size, void* d_ws, size_t ws_size,
                              hipStream_t stream)
{
    const float* k    = (const float*)d_in[0];   // [8,2048,1024]
    const float* q    = (const float*)d_in[1];   // [8,2048,1024]
    const float* Wk   = (const float*)d_in[2];   // kx = k @ Wk^T + bk
    const float* bk   = (const float*)d_in[3];
    const float* Wq   = (const float*)d_in[4];
    const float* bq   = (const float*)d_in[5];
    const float* Wbil = (const float*)d_in[6];
    const float* Wp   = (const float*)d_in[7];
    const float* bp   = (const float*)d_in[8];

    const int Bn = 8, L = 2048, D = 1024;
    const long long MD = (long long)Bn * L;      // 16384 rows

    float* out   = (float*)d_out;                // [16384,1024]
    float* score = (float*)d_out + MD * D;       // [8,2048,2048]

    float* ws = (float*)d_ws;
    float* kx = ws;                              // [16384,1024]
    float* qw = ws + MD * D;                     // [16384,1024], reused as out_mid
    float* Wc = ws + 2 * MD * D;                 // [1024,1024] = Wq^T @ W_bil
    float* bc = Wc + (long long)D * D;           // [1024]      = bq @ W_bil

    // (1) Wc = Wq^T @ W_bil  (A transposed view, B NN)
    launch_gemm<false,false>(Wq, 1, D, 0,  Wbil, 1, D, 0,  Wc, 0, nullptr, D, D, D, 1, stream);
    // (2) bc = bq @ W_bil
    bil_bias<<<dim3(D / 256), 256, 0, stream>>>(bq, Wbil, bc, D);
    // (3) kx = k @ Wk^T + bk  (NT)
    launch_gemm<true,true>(k, D, 1, 0,  Wk, D, 1, 0,  kx, 0, bk, (int)MD, D, D, 1, stream);
    // (4) qw = q @ Wc + bc   (fused q-projection+bilinear, NN)
    launch_gemm<true,false>(q, D, 1, 0,  Wc, 1, D, 0,  qw, 0, bc, (int)MD, D, D, 1, stream);
    // (5) logits[b] = qw[b] @ kx[b]^T  (batched NT) -> score region
    launch_gemm<true,true>(qw, D, 1, (long long)L * D,
                           kx, D, 1, (long long)L * D,
                           score, (long long)L * L, nullptr, L, L, D, Bn, stream);
    // (6) softmax rows in place (score is a required f32 output)
    softmax_rows<<<dim3((unsigned)MD), 256, 0, stream>>>(score, L);
    // (7) out_mid[b] = score[b] @ kx[b]  (batched NN), reuse qw buffer
    launch_gemm<true,false>(score, L, 1, (long long)L * L,
                            kx, 1, D, (long long)L * D,
                            qw, (long long)L * D, nullptr, L, D, L, Bn, stream);
    // (8) out = out_mid @ Wp^T + bp  (NT)
    launch_gemm<true,true>(qw, D, 1, 0,  Wp, D, 1, 0,  out, 0, bp, (int)MD, D, D, 1, stream);
}
